// GCN_22308060136220
// MI455X (gfx1250) — matrix operand order repaired and checked by reading, not compiled, not run on hardware
//
#include <hip/hip_runtime.h>
#include <hip/hip_bf16.h>

// ---------------------------------------------------------------------------
// GCN (4x GCNConv 128->128 + head 128->64) for gfx1250 (MI455X).
// L2-resident graph aggregation (wave-per-edge f32 atomic scatter), bf16 WMMA
// GEMMs with f32 accumulation, degree normalization computed once. The bf16
// down-convert is fused into the finalize pass so each layer's activations
// are written exactly once.
// ---------------------------------------------------------------------------

typedef __attribute__((ext_vector_type(16))) __bf16 v16bf;
typedef __attribute__((ext_vector_type(8)))  __bf16 v8bf;
typedef __attribute__((ext_vector_type(4)))  __bf16 v4bf;
typedef __attribute__((ext_vector_type(8)))  float  v8f;

__device__ __forceinline__ void atomAddF32(float* p, float v) {
    __hip_atomic_fetch_add(p, v, __ATOMIC_RELAXED, __HIP_MEMORY_SCOPE_AGENT);
}

// ---------------- fp32 -> bf16 conversion (only for the input x) -----------
__global__ void convert_bf16_kernel(const float* __restrict__ src,
                                    __bf16* __restrict__ dst, long long n4) {
    long long i = (long long)blockIdx.x * blockDim.x + threadIdx.x;
    if (i >= n4) return;
    float4 v = ((const float4*)src)[i];
    v4bf o;
    o[0] = (__bf16)v.x; o[1] = (__bf16)v.y; o[2] = (__bf16)v.z; o[3] = (__bf16)v.w;
    ((v4bf*)dst)[i] = o;
}

// -------- fp32 W[K][N] -> bf16 Wt[N][K] (transpose so B frags are contiguous)
__global__ void convert_transpose_w_kernel(const float* __restrict__ W,
                                           __bf16* __restrict__ Wt,
                                           int K, int Nc) {
    int id = blockIdx.x * blockDim.x + threadIdx.x;
    if (id >= K * Nc) return;
    int k = id / Nc;
    int n = id - k * Nc;
    Wt[(long long)n * K + k] = (__bf16)W[(long long)k * Nc + n];
}

// ---------------- degree accumulation + inverse sqrt -----------------------
__global__ void degree_kernel(const long long* __restrict__ ei,
                              float* __restrict__ deg, int E) {
    int e = blockIdx.x * blockDim.x + threadIdx.x;
    if (e >= E) return;
    long long d = __builtin_nontemporal_load(ei + E + e);   // streamed once
    atomAddF32(&deg[d], 1.0f);
}

__global__ void invsqrt_kernel(float* __restrict__ deg, int n) {
    int i = blockIdx.x * blockDim.x + threadIdx.x;
    if (i >= n) return;
    float dv = deg[i] + 1.0f;  // + self loop; always >= 1
    deg[i] = rsqrtf(dv);       // in place: buffer becomes inv_sqrt(deg)
}

// ---------------- WMMA fragment loaders ------------------------------------
// A (16x32 bf16, MxK): lane<16 holds row M=lane, elems 0..7 = K0..7,
// elems 8..15 = K16..23; lane>=16 holds row M=lane-16, K8..15 / K24..31.
__device__ __forceinline__ v16bf load_frag_a(const __bf16* __restrict__ rowBase,
                                             int k0, int half) {
    const int off = half * 8;
    v8bf lo = *(const v8bf*)(rowBase + k0 + off);
    v8bf hi = *(const v8bf*)(rowBase + k0 + 16 + off);
    return __builtin_shufflevector(lo, hi, 0, 1, 2, 3, 4, 5, 6, 7,
                                   8, 9, 10, 11, 12, 13, 14, 15);
}

// B (32x16 bf16, KxN), fed from transposed W (Wt[N][K], row-major in K):
// lane<16 holds col N=lane, K0..7 / K16..23; lane>=16 holds col, K8..15 / K24..31.
__device__ __forceinline__ v16bf load_frag_b(const __bf16* __restrict__ colBase,
                                             int k0, int half) {
    const int off = half * 8;
    v8bf lo = *(const v8bf*)(colBase + k0 + off);
    v8bf hi = *(const v8bf*)(colBase + k0 + 16 + off);
    return __builtin_shufflevector(lo, hi, 0, 1, 2, 3, 4, 5, 6, 7,
                                   8, 9, 10, 11, 12, 13, 14, 15);
}

// ---------------- bf16 WMMA GEMM: C[M][N] = A[M][128] * Wt[N][128]^T + bias -
// One wave computes a 16 x (NT*16) strip. K fixed at 128 (4 steps of 32).
// M must be a multiple of 16 (50000 = 3125*16). EXEC stays all-ones per wave.
template <int NT>
__global__ void gemm_wmma_kernel(const __bf16* __restrict__ A,
                                 const __bf16* __restrict__ Bt,
                                 const float* __restrict__ bias,
                                 float* __restrict__ C,
                                 int M, int N) {
    const int lane = threadIdx.x & 31;
    const int half = lane >> 4;   // 0: lanes 0-15, 1: lanes 16-31
    const int lid  = lane & 15;
    const int wavesPerBlock = blockDim.x >> 5;
    const int tile = blockIdx.x * wavesPerBlock + (threadIdx.x >> 5);
    const int tilesM = M >> 4;
    if (tile >= tilesM) return;           // wave-uniform exit
    const int row0 = tile << 4;

    const __bf16* __restrict__ arow = A + (long long)(row0 + lid) * 128;

    v8f zero = {0.f, 0.f, 0.f, 0.f, 0.f, 0.f, 0.f, 0.f};
    v8f acc[NT];
#pragma unroll
    for (int j = 0; j < NT; ++j) acc[j] = zero;

#pragma unroll
    for (int ks = 0; ks < 4; ++ks) {
        const int k0 = ks * 32;
        __builtin_prefetch(arow + k0 + 32, 0, 0);   // global_prefetch_b8
        v16bf afrag = load_frag_a(arow, k0, half);
#pragma unroll
        for (int j = 0; j < NT; ++j) {
            const __bf16* __restrict__ bcol = Bt + (long long)(j * 16 + lid) * 128;
            v16bf bfrag = load_frag_b(bcol, k0, half);
            acc[j] = __builtin_amdgcn_wmma_f32_16x16x32_bf16(
                false, afrag, false, bfrag, (short)0, acc[j], false, false);
        }
    }

    // C/D layout: VGPR r -> row = row0 + r + 8*half, col = col0 + lid.
#pragma unroll
    for (int j = 0; j < NT; ++j) {
        const int col = j * 16 + lid;
        const float badd = bias ? bias[col] : 0.0f;
#pragma unroll
        for (int r = 0; r < 8; ++r) {
            const int row = row0 + r + half * 8;
            C[(long long)row * N + col] = acc[j][r] + badd;
        }
    }
}

// ---------------- wave-per-edge gather/scale/scatter-add --------------------
// Lane l handles features [4l, 4l+4): a float4 load of t[src], scale by
// coef = inv_sqrt[src]*inv_sqrt[dst], four f32 atomic adds into agg[dst].
__global__ void scatter_edges_kernel(const float* __restrict__ t,
                                     const long long* __restrict__ ei,
                                     const float* __restrict__ invs,
                                     float* __restrict__ agg, int E) {
    const int lane = threadIdx.x & 31;
    const long long e = (long long)blockIdx.x * (blockDim.x >> 5) + (threadIdx.x >> 5);
    if (e >= E) return;                   // wave-uniform exit
    const long long s = __builtin_nontemporal_load(ei + e);      // streamed once
    const long long d = __builtin_nontemporal_load(ei + E + e);  // streamed once
    const float coef = invs[s] * invs[d];
    const float4 v = *(const float4*)(t + s * 128 + lane * 4);   // L2-hot reuse
    float* dst = agg + d * 128 + lane * 4;
    atomAddF32(dst + 0, v.x * coef);
    atomAddF32(dst + 1, v.y * coef);
    atomAddF32(dst + 2, v.z * coef);
    atomAddF32(dst + 3, v.w * coef);
}

// ------- finalize: agg + self-loop + bias (+ ReLU), emit bf16 directly -----
__global__ void finalize_kernel(const float* __restrict__ agg,
                                const float* __restrict__ t,
                                const float* __restrict__ invs,
                                const float* __restrict__ bias,
                                __bf16* __restrict__ out_bf,
                                long long nodes, int relu) {
    long long i = (long long)blockIdx.x * blockDim.x + threadIdx.x;
    if (i >= nodes * 32) return;          // float4 / v4bf elements
    const long long node = i >> 5;
    const int q = (int)(i & 31);
    const float iv = invs[node];
    const float sg = iv * iv;             // self-loop coefficient 1/deg
    const float4 a  = ((const float4*)agg)[i];
    const float4 tv = ((const float4*)t)[i];
    const float4 bv = ((const float4*)bias)[q];
    float4 r;
    r.x = a.x + tv.x * sg + bv.x;
    r.y = a.y + tv.y * sg + bv.y;
    r.z = a.z + tv.z * sg + bv.z;
    r.w = a.w + tv.w * sg + bv.w;
    if (relu) {
        r.x = fmaxf(r.x, 0.f); r.y = fmaxf(r.y, 0.f);
        r.z = fmaxf(r.z, 0.f); r.w = fmaxf(r.w, 0.f);
    }
    v4bf o;
    o[0] = (__bf16)r.x; o[1] = (__bf16)r.y; o[2] = (__bf16)r.z; o[3] = (__bf16)r.w;
    ((v4bf*)out_bf)[i] = o;               // next GEMM's A operand, written once
}

// ---------------------------------------------------------------------------
extern "C" void kernel_launch(void* const* d_in, const int* in_sizes, int n_in,
                              void* d_out, int out_size, void* d_ws, size_t ws_size,
                              hipStream_t stream) {
    const float*     x  = (const float*)d_in[0];
    const long long* ei = (const long long*)d_in[1];
    const float* W[4] = {(const float*)d_in[2], (const float*)d_in[4],
                         (const float*)d_in[6], (const float*)d_in[8]};
    const float* b[4] = {(const float*)d_in[3], (const float*)d_in[5],
                         (const float*)d_in[7], (const float*)d_in[9]};
    const float* Wh = (const float*)d_in[10];
    const float* bh = (const float*)d_in[11];

    const int DIM  = 128;
    const int HOUT = 64;
    const long long nodes = in_sizes[0] / DIM;      // 50000
    const int E = in_sizes[1] / 2;                  // 800000
    const long long feat = nodes * DIM;             // node-feature elements

    // ---- workspace carve-up (256B aligned) ----
    char* ws = (char*)d_ws;
    size_t off = 0;
    auto carve = [&](size_t bytes) {
        void* p = ws + off;
        off += (bytes + 255) & ~(size_t)255;
        return p;
    };
    float*  invs = (float*)carve(nodes * sizeof(float));           // deg -> inv_sqrt
    float*  t    = (float*)carve(feat * sizeof(float));            // GEMM output
    float*  agg  = (float*)carve(feat * sizeof(float));            // scatter target
    __bf16* h_bf = (__bf16*)carve(feat * sizeof(__bf16));          // activations (bf16)
    __bf16* wt   = (__bf16*)carve((size_t)DIM * DIM * sizeof(__bf16)); // bf16 Wt
    (void)ws_size; (void)n_in; (void)out_size;

    const int TPB = 256;

    // ---- degrees + symmetric normalization (graph fixed across layers) ----
    hipMemsetAsync(invs, 0, nodes * sizeof(float), stream);
    degree_kernel<<<(E + TPB - 1) / TPB, TPB, 0, stream>>>(ei, invs, E);
    invsqrt_kernel<<<((int)nodes + TPB - 1) / TPB, TPB, 0, stream>>>(invs, (int)nodes);

    const long long n4 = feat / 4;
    const int convBlocks = (int)((n4 + TPB - 1) / TPB);
    const int tilesM = (int)(nodes / 16);           // 3125, exact
    const int gemmBlocks = (tilesM + (TPB / 32) - 1) / (TPB / 32);
    const int scatterBlocks = (E + (TPB / 32) - 1) / (TPB / 32);
    const int finBlocks = (int)((nodes * 32 + TPB - 1) / TPB);

    // initial activations: x -> bf16 (only explicit conversion pass)
    convert_bf16_kernel<<<convBlocks, TPB, 0, stream>>>(x, h_bf, n4);

    for (int l = 0; l < 4; ++l) {
        convert_transpose_w_kernel<<<(DIM * DIM + TPB - 1) / TPB, TPB, 0, stream>>>(
            W[l], wt, DIM, DIM);
        gemm_wmma_kernel<8><<<gemmBlocks, TPB, 0, stream>>>(
            h_bf, wt, nullptr, t, (int)nodes, DIM);
        hipMemsetAsync(agg, 0, feat * sizeof(float), stream);
        scatter_edges_kernel<<<scatterBlocks, TPB, 0, stream>>>(t, ei, invs, agg, E);
        finalize_kernel<<<finBlocks, TPB, 0, stream>>>(
            agg, t, invs, b[l], h_bf, nodes, (l < 3) ? 1 : 0);
    }

    // ---- head: out = h @ Wh + bh (no graph conv) ----
    convert_transpose_w_kernel<<<(DIM * HOUT + TPB - 1) / TPB, TPB, 0, stream>>>(
        Wh, wt, DIM, HOUT);
    gemm_wmma_kernel<4><<<gemmBlocks, TPB, 0, stream>>>(
        h_bf, wt, bh, (float*)d_out, (int)nodes, HOUT);
}
